// FormerDecoderAttention_81303730913415
// MI455X (gfx1250) — matrix operand order repaired and checked by reading, not compiled, not run on hardware
//
#include <hip/hip_runtime.h>
#include <hip/hip_bf16.h>

// ---------------------------------------------------------------------------
// Problem constants (from reference)
// ---------------------------------------------------------------------------
#define BB   8
#define CC   512
#define LL   2048
#define HH   8
#define KK   3
#define DD   64          // head dim
#define LQ   2046        // conv output length
#define LQV  2048        // padded key stride for V (keeps b128 loads in-bounds)

typedef __attribute__((ext_vector_type(16))) __bf16 v16bf;
typedef __attribute__((ext_vector_type(8)))  float  v8f;

union B32x16 { uint4 u[2]; v16bf v; };   // 32 bytes = one bf16 WMMA operand per lane

__device__ __forceinline__ unsigned short f2bf(float f) {
    unsigned int x = __float_as_uint(f);
    unsigned int r = x + 0x7FFFu + ((x >> 16) & 1u);   // round-to-nearest-even
    return (unsigned short)(r >> 16);
}

// ---------------------------------------------------------------------------
// Kernel 1: grouped-conv QKV projections (f32 math, bf16 outputs)
//   Q,K stored [B,H,LQ,64]  (row-major over d)  -> WMMA A / B operands for S
//   V   stored [B,H,64,2048] (row-major over t) -> WMMA B operand for PV
// ---------------------------------------------------------------------------
__global__ __launch_bounds__(256) void conv_qkv_kernel(
    const float* __restrict__ in, const float* __restrict__ qw,
    const float* __restrict__ kw, const float* __restrict__ vw,
    unsigned short* __restrict__ Qb, unsigned short* __restrict__ Kb,
    unsigned short* __restrict__ Vb)
{
    __shared__ float xs[64 * 66];
    const int tblk = blockIdx.x, g = blockIdx.y, b = blockIdx.z;
    const int t0 = tblk * 64;

    for (int idx = threadIdx.x; idx < 64 * 66; idx += 256) {
        int i = idx / 66, tt = idx % 66;
        int gt = t0 + tt;
        xs[idx] = (gt < LL) ? in[((size_t)b * CC + g * 64 + i) * LL + gt] : 0.0f;
    }
    __syncthreads();

    const int o  = threadIdx.x >> 2;   // 0..63 output channel in group
    const int ts = threadIdx.x & 3;    // 0..3  sixteen-t slot
    const int wbase = (g * 64 + o) * (64 * KK);

    float aq[16], ak[16], av[16];
#pragma unroll
    for (int n = 0; n < 16; ++n) { aq[n] = 0.f; ak[n] = 0.f; av[n] = 0.f; }

    for (int i = 0; i < 64; ++i) {
        float x[18];
#pragma unroll
        for (int n = 0; n < 18; ++n) x[n] = xs[i * 66 + ts * 16 + n];
#pragma unroll
        for (int kkk = 0; kkk < KK; ++kkk) {
            float wq_ = qw[wbase + i * KK + kkk];
            float wk_ = kw[wbase + i * KK + kkk];
            float wv_ = vw[wbase + i * KK + kkk];
#pragma unroll
            for (int n = 0; n < 16; ++n) {
                aq[n] = fmaf(wq_, x[n + kkk], aq[n]);
                ak[n] = fmaf(wk_, x[n + kkk], ak[n]);
                av[n] = fmaf(wv_, x[n + kkk], av[n]);
            }
        }
    }

#pragma unroll
    for (int n = 0; n < 16; ++n) {
        int t = t0 + ts * 16 + n;
        if (t < LQ) {
            size_t qk_idx = (((size_t)b * HH + g) * LQ + t) * DD + o;
            Qb[qk_idx] = f2bf(aq[n]);
            Kb[qk_idx] = f2bf(ak[n]);
            Vb[(((size_t)b * HH + g) * DD + o) * LQV + t] = f2bf(av[n]);
        }
    }
}

// ---------------------------------------------------------------------------
// Kernel 2: fused causal attention (flash-style, bf16 WMMA, f32 softmax)
// 4 waves / block; each wave owns a 16-query tile of one (b,h).
// ---------------------------------------------------------------------------
__global__ __launch_bounds__(128) void attn_kernel(
    const unsigned short* __restrict__ Qb, const unsigned short* __restrict__ Kb,
    const unsigned short* __restrict__ Vb, float* __restrict__ attn_out)
{
    __shared__ __align__(16) unsigned short pbuf[4][16][32];  // per-wave P staging

    const int lane = threadIdx.x & 31;
    const int w    = threadIdx.x >> 5;
    const int h    = blockIdx.y, b = blockIdx.z;
    const int qt   = blockIdx.x * 4 + w;
    const int qbase = qt * 16;

    const int half = lane >> 4;   // 0 or 1
    const int lm   = lane & 15;

    const unsigned short* Qh = Qb + ((size_t)(b * HH + h)) * LQ * DD;
    const unsigned short* Kh = Kb + ((size_t)(b * HH + h)) * LQ * DD;
    const unsigned short* Vh = Vb + ((size_t)(b * HH + h)) * DD * LQV;

    // --- A operand: Q tile, 16(q) x 64(d) as two 16x32 bf16 operands --------
    B32x16 aq[2];
    {
        int r = min(qbase + lm, LQ - 1);
#pragma unroll
        for (int c = 0; c < 2; ++c) {
            aq[c].u[0] = *(const uint4*)(Qh + (size_t)r * DD + c * 32 + half * 8);
            aq[c].u[1] = *(const uint4*)(Qh + (size_t)r * DD + c * 32 + 16 + half * 8);
        }
    }

    v8f ot[4] = {};                    // O accumulator: 16q x 64d (4 d-tiles)
    float m[8], l[8];
#pragma unroll
    for (int j = 0; j < 8; ++j) { m[j] = -1.0e30f; l[j] = 0.0f; }

    const int   qmax = min(qbase + 15, LQ - 1);
    const float sc   = 0.35355339059327379f;   // 1/sqrt(H=8)

    for (int kt = 0; kt <= qmax; kt += 32) {
        // ---- S = (Q^T K) tile: 16q x 32k as two 16x16 f32 D-tiles ----------
        v8f s[2] = {};
#pragma unroll
        for (int nt = 0; nt < 2; ++nt) {
            int key = min(kt + nt * 16 + lm, LQ - 1);
#pragma unroll
            for (int c = 0; c < 2; ++c) {
                B32x16 bk;
                bk.u[0] = *(const uint4*)(Kh + (size_t)key * DD + c * 32 + half * 8);
                bk.u[1] = *(const uint4*)(Kh + (size_t)key * DD + c * 32 + 16 + half * 8);
                s[nt] = __builtin_amdgcn_wmma_f32_16x16x32_bf16(
                    false, aq[c].v, false, bk.v, (short)0, s[nt], false, false);
            }
        }

        // ---- scale + causal mask ------------------------------------------
        const bool needmask = (kt + 31 > qbase);
#pragma unroll
        for (int nt = 0; nt < 2; ++nt) {
#pragma unroll
            for (int j = 0; j < 8; ++j) {
                float v = s[nt][j] * sc;
                if (needmask) {
                    int q   = qbase + j + half * 8;
                    int col = kt + nt * 16 + lm;
                    v = (col > q) ? v - 1.0e9f : v;
                }
                s[nt][j] = v;
            }
        }

        // ---- online softmax (per-row; rows live in 16-lane halves) --------
#pragma unroll
        for (int j = 0; j < 8; ++j) {
            float mt = fmaxf(s[0][j], s[1][j]);
#pragma unroll
            for (int d = 8; d >= 1; d >>= 1) mt = fmaxf(mt, __shfl_xor(mt, d, 32));
            float mn  = fmaxf(m[j], mt);
            float scl = __expf(m[j] - mn);
            float p0  = __expf(s[0][j] - mn);
            float p1  = __expf(s[1][j] - mn);
            float rs  = p0 + p1;
#pragma unroll
            for (int d = 8; d >= 1; d >>= 1) rs += __shfl_xor(rs, d, 32);
            l[j] = l[j] * scl + rs;
            m[j] = mn;
            ot[0][j] *= scl; ot[1][j] *= scl; ot[2][j] *= scl; ot[3][j] *= scl;

            int row = j + half * 8;
            pbuf[w][row][lm]      = f2bf(p0);
            pbuf[w][row][16 + lm] = f2bf(p1);
        }

        // wave-local LDS fence: drain ds stores before re-reading transposed
        asm volatile("s_wait_dscnt 0" ::: "memory");

        // ---- re-read P as 16x32 bf16 A operand ----------------------------
        B32x16 ap;
        ap.u[0] = *(const uint4*)&pbuf[w][lm][half * 8];
        ap.u[1] = *(const uint4*)&pbuf[w][lm][16 + half * 8];
        asm volatile("s_wait_dscnt 0" ::: "memory");

        // ---- O += P * V ----------------------------------------------------
#pragma unroll
        for (int dt = 0; dt < 4; ++dt) {
            int dd = dt * 16 + lm;
            B32x16 bv;
            bv.u[0] = *(const uint4*)(Vh + (size_t)dd * LQV + kt + half * 8);
            bv.u[1] = *(const uint4*)(Vh + (size_t)dd * LQV + kt + 16 + half * 8);
            ot[dt] = __builtin_amdgcn_wmma_f32_16x16x32_bf16(
                false, ap.v, false, bv.v, (short)0, ot[dt], false, false);
        }
    }

    // ---- normalize + store (attn_out is [B, LQ, C], head-major channels) --
#pragma unroll
    for (int dt = 0; dt < 4; ++dt) {
#pragma unroll
        for (int j = 0; j < 8; ++j) {
            int q = qbase + j + half * 8;
            if (q < LQ) {
                int c = h * DD + dt * 16 + lm;
                attn_out[((size_t)b * LQ + q) * CC + c] = ot[dt][j] / l[j];
            }
        }
    }
}

// ---------------------------------------------------------------------------
// Kernel 3: residual add + LayerNorm over C=512, one block per (b, q)
// ---------------------------------------------------------------------------
__global__ __launch_bounds__(256) void ln_kernel(
    const float* __restrict__ attn, const float* __restrict__ initw,
    const float* __restrict__ gamma, const float* __restrict__ beta,
    float* __restrict__ out)
{
    __shared__ float rs[8], rq[8];
    const int q = blockIdx.x, b = blockIdx.y;
    const int c0 = threadIdx.x, c1 = threadIdx.x + 256;

    size_t rowo = ((size_t)b * LQ + q) * CC;
    float x0 = attn[rowo + c0] + initw[((size_t)b * CC + c0) * LQ + q];
    float x1 = attn[rowo + c1] + initw[((size_t)b * CC + c1) * LQ + q];

    float s  = x0 + x1;
    float ss = x0 * x0 + x1 * x1;
#pragma unroll
    for (int d = 16; d >= 1; d >>= 1) {
        s  += __shfl_xor(s,  d, 32);
        ss += __shfl_xor(ss, d, 32);
    }
    int lane = threadIdx.x & 31, wv = threadIdx.x >> 5;
    if (lane == 0) { rs[wv] = s; rq[wv] = ss; }
    __syncthreads();
    float ts = 0.f, tq = 0.f;
#pragma unroll
    for (int i = 0; i < 8; ++i) { ts += rs[i]; tq += rq[i]; }

    float mu   = ts * (1.0f / CC);
    float var  = tq * (1.0f / CC) - mu * mu;
    float rstd = rsqrtf(var + 1e-5f);

    out[rowo + c0] = (x0 - mu) * rstd * gamma[c0] + beta[c0];
    out[rowo + c1] = (x1 - mu) * rstd * gamma[c1] + beta[c1];
}

// ---------------------------------------------------------------------------
// Launch
// ---------------------------------------------------------------------------
extern "C" void kernel_launch(void* const* d_in, const int* in_sizes, int n_in,
                              void* d_out, int out_size, void* d_ws, size_t ws_size,
                              hipStream_t stream) {
    const float* in    = (const float*)d_in[0];
    const float* initw = (const float*)d_in[1];
    const float* qw    = (const float*)d_in[2];
    const float* kw    = (const float*)d_in[3];
    const float* vw    = (const float*)d_in[4];
    const float* gamma = (const float*)d_in[5];
    const float* beta  = (const float*)d_in[6];
    // d_in[7] = mask; causal triu(k=1) computed analytically in-kernel.

    // Workspace layout (bytes): Q,K bf16 [B,H,LQ,64]; V bf16 [B,H,64,2048]; attn f32 [B,LQ,C]
    const size_t SQ = (size_t)BB * HH * LQ * DD * 2;   // 16,760,832
    const size_t SV = (size_t)BB * HH * DD * LQV * 2;  // 16,777,216
    unsigned short* Qb = (unsigned short*)d_ws;
    unsigned short* Kb = (unsigned short*)((char*)d_ws + SQ);
    unsigned short* Vb = (unsigned short*)((char*)d_ws + 2 * SQ);
    float* attn        = (float*)((char*)d_ws + 2 * SQ + SV);
    float* out         = (float*)d_out;

    conv_qkv_kernel<<<dim3(32, HH, BB), 256, 0, stream>>>(in, qw, kw, vw, Qb, Kb, Vb);
    attn_kernel<<<dim3(32, HH, BB), 128, 0, stream>>>(Qb, Kb, Vb, attn);
    ln_kernel<<<dim3(LQ, BB), 256, 0, stream>>>(attn, initw, gamma, beta, out);
}
